// FeaturePropagation_30657476559328
// MI455X (gfx1250) — matrix operand order repaired and checked by reading, not compiled, not run on hardware
//
#include <hip/hip_runtime.h>
#include <stdint.h>

typedef __attribute__((ext_vector_type(8)))  unsigned short us8;
typedef __attribute__((ext_vector_type(16))) unsigned short us16;
typedef __attribute__((ext_vector_type(16))) __bf16        v16bf;
typedef __attribute__((ext_vector_type(8)))  float         v8f;
typedef __attribute__((ext_vector_type(4)))  int           i32x4;

typedef __attribute__((address_space(1))) i32x4* gvec_p;
typedef __attribute__((address_space(3))) i32x4* lvec_p;

#define B_    8
#define N1_   16384
#define N2_   2048
#define C1_   128
#define C2_   256
#define INCH_ 384
#define M_    (B_ * N1_)   // 131072 rows

#if defined(__has_builtin)
#if __has_builtin(__builtin_amdgcn_global_load_async_to_lds_b128) && \
    __has_builtin(__builtin_amdgcn_s_wait_asynccnt)
#define HAVE_ASYNC_LDS 1
#endif
#endif

__device__ __forceinline__ unsigned short f2bf(float f) {
  unsigned u = __float_as_uint(f);
  u += 0x7FFFu + ((u >> 16) & 1u);       // round-to-nearest-even
  return (unsigned short)(u >> 16);
}

// ---------------- weight f32 -> bf16 ----------------
__global__ void cvt_f32_bf16(const float* __restrict__ in,
                             unsigned short* __restrict__ out, int n) {
  int i = blockIdx.x * blockDim.x + threadIdx.x;
  if (i < n) out[i] = f2bf(in[i]);
}

// ---------------- 3-NN + inverse-distance weights ----------------
// block: 256 threads, grid (N1/256, B). xyz2 tile (24KB) cached in LDS.
__global__ void knn3_kernel(const float* __restrict__ xyz1,
                            const float* __restrict__ xyz2,
                            int* __restrict__ idx_out,
                            float* __restrict__ w_out) {
  __shared__ float s2[N2_ * 3];
  int b = blockIdx.y;
  const float* x2 = xyz2 + (size_t)b * N2_ * 3;
  for (int j = threadIdx.x; j < N2_ * 3; j += blockDim.x) s2[j] = x2[j];
  __syncthreads();

  int n = blockIdx.x * blockDim.x + threadIdx.x;
  size_t row = (size_t)b * N1_ + n;
  float px = xyz1[row * 3 + 0], py = xyz1[row * 3 + 1], pz = xyz1[row * 3 + 2];

  float d0 = 3.4e38f, d1 = 3.4e38f, d2 = 3.4e38f;
  int   i0 = 0, i1 = 0, i2 = 0;
  for (int j = 0; j < N2_; ++j) {                 // LDS broadcast reads
    float dx = px - s2[3 * j + 0];
    float dy = py - s2[3 * j + 1];
    float dz = pz - s2[3 * j + 2];
    float d = dx * dx + dy * dy + dz * dz;
    if (d < d2) {
      if (d < d1) {
        d2 = d1; i2 = i1;
        if (d < d0) { d1 = d0; i1 = i0; d0 = d; i0 = j; }
        else        { d1 = d;  i1 = j; }
      } else { d2 = d; i2 = j; }
    }
  }
  d0 = fmaxf(d0, 1e-10f); d1 = fmaxf(d1, 1e-10f); d2 = fmaxf(d2, 1e-10f);
  float w0 = 1.f / d0, w1 = 1.f / d1, w2 = 1.f / d2;
  float rs = 1.f / fmaxf(w0 + w1 + w2, 1e-8f);
  idx_out[row * 3 + 0] = i0; idx_out[row * 3 + 1] = i1; idx_out[row * 3 + 2] = i2;
  w_out[row * 3 + 0] = w0 * rs; w_out[row * 3 + 1] = w1 * rs; w_out[row * 3 + 2] = w2 * rs;
}

// ---------------- interpolate + concat -> bf16 activation matrix ----------------
// block: 128 threads, one row per block. xbf[row][0:128]=points1, [128:384]=interp.
__global__ void interp_concat_kernel(const float* __restrict__ points1,
                                     const float* __restrict__ points2,
                                     const int* __restrict__ idx3,
                                     const float* __restrict__ w3,
                                     unsigned short* __restrict__ xbf) {
  size_t row = blockIdx.x;
  int b = (int)(row >> 14);                       // row / N1
  int tid = threadIdx.x;
  int   i0 = idx3[row * 3 + 0], i1 = idx3[row * 3 + 1], i2 = idx3[row * 3 + 2];
  float w0 = w3[row * 3 + 0],  w1 = w3[row * 3 + 1],  w2 = w3[row * 3 + 2];
  const float* p1 = points1 + row * C1_;
  const float* p2 = points2 + (size_t)b * N2_ * C2_;
  unsigned short* xr = xbf + row * INCH_;
  xr[tid] = f2bf(p1[tid]);                        // tid in [0,128)
  for (int c = tid; c < C2_; c += 128) {
    float v = w0 * p2[(size_t)i0 * C2_ + c]
            + w1 * p2[(size_t)i1 * C2_ + c]
            + w2 * p2[(size_t)i2 * C2_ + c];
    xr[C1_ + c] = f2bf(v);
  }
}

// ---------------- bf16 WMMA GEMM + bias + BN-stat accumulation ----------------
// A: [M x K] bf16 row-major.  Bw: [N x K] bf16 row-major (== W as given).
// B tile (64 x K) for this block's n-range is staged into LDS once via the
// async-to-LDS path (ASYNCcnt) and reused by all 8 waves; A streams from
// global (L2-resident). Also accumulates per-channel sum/sumsq for batch-norm.
// block = 256 threads (8 waves); wave w -> rows m0 = bx*128 + w*16, cols n0 = by*64.
template <int K, int N>
__global__ __launch_bounds__(256)
void gemm_bf16_wmma(const unsigned short* __restrict__ A,
                    const unsigned short* __restrict__ Bw,
                    const float* __restrict__ bias,
                    float* __restrict__ C,
                    float* __restrict__ gsum, float* __restrict__ gsq) {
  constexpr int LDB = K + 8;   // padded row stride (ushorts): keeps 16B align, kills bank conflicts
  __shared__ unsigned short sB[64 * LDB];
  __shared__ float lsum[64], lsq[64];
  int tid = threadIdx.x;
  if (tid < 64) { lsum[tid] = 0.f; lsq[tid] = 0.f; }

  int n0 = blockIdx.y * 64;

  // ---- stage B tile (64 rows x K) into LDS, 16B chunks, async-to-LDS path ----
  constexpr int CPR = K / 8;                 // us8 chunks per row
  for (int c = tid; c < 64 * CPR; c += 256) {
    int row = c / CPR, col = (c % CPR) * 8;
    const unsigned short* gp = Bw + (size_t)(n0 + row) * K + col;
    unsigned short* lp = &sB[row * LDB + col];
#ifdef HAVE_ASYNC_LDS
    __builtin_amdgcn_global_load_async_to_lds_b128(
        (gvec_p)(void*)gp, (lvec_p)(void*)lp, 0, 0);
#else
    *(us8*)lp = *(const us8*)gp;
#endif
  }
#ifdef HAVE_ASYNC_LDS
  __builtin_amdgcn_s_wait_asynccnt(0);
#endif
  __syncthreads();

  int wave = tid >> 5, lane = tid & 31;
  int half = lane >> 4, lm = lane & 15;
  int m0 = blockIdx.x * 128 + wave * 16;

  const unsigned short* arow = A + (size_t)(m0 + lm) * K;
  v8f acc[4] = {};

#pragma unroll
  for (int kb = 0; kb < K; kb += 32) {
    // A fragment (16x32 bf16): elems 0-7 = K kb+half*8.., elems 8-15 = K kb+16+half*8..
    us8 alo = *(const us8*)(arow + kb + half * 8);
    us8 ahi = *(const us8*)(arow + kb + 16 + half * 8);
    us16 au;
#pragma unroll
    for (int i = 0; i < 8; ++i) { au[i] = alo[i]; au[8 + i] = ahi[i]; }
    v16bf afrag = __builtin_bit_cast(v16bf, au);
#pragma unroll
    for (int nt = 0; nt < 4; ++nt) {
      // B fragment (32x16): lane holds col n=lm, elems 0-7 = K kb+half*8.., elems 8-15 = K kb+16+half*8..
      const unsigned short* brow = &sB[(nt * 16 + lm) * LDB + kb + half * 8];
      us8 blo = *(const us8*)(brow);
      us8 bhi = *(const us8*)(brow + 16);
      us16 bu;
#pragma unroll
      for (int i = 0; i < 8; ++i) { bu[i] = blo[i]; bu[8 + i] = bhi[i]; }
      v16bf bfrag = __builtin_bit_cast(v16bf, bu);
      acc[nt] = __builtin_amdgcn_wmma_f32_16x16x32_bf16(
          false, afrag, false, bfrag, (short)0, acc[nt], false, false);
    }
  }

#pragma unroll
  for (int nt = 0; nt < 4; ++nt) {
    int n = n0 + nt * 16 + lm;
    float bn = bias[n];
    float s = 0.f, s2 = 0.f;
#pragma unroll
    for (int r = 0; r < 8; ++r) {               // C/D: lane=col n, rows half*8+r
      float v = acc[nt][r] + bn;
      C[(size_t)(m0 + half * 8 + r) * N + n] = v;
      s += v; s2 += v * v;
    }
    atomicAdd(&lsum[nt * 16 + lm], s);
    atomicAdd(&lsq [nt * 16 + lm], s2);
  }
  __syncthreads();
  if (tid < 64) {
    atomicAdd(&gsum[n0 + tid], lsum[tid]);
    atomicAdd(&gsq [n0 + tid], lsq [tid]);
  }
}

// ---------------- BN (training-mode batch stats) + ReLU ----------------
template <bool BF16OUT>
__global__ void bnrelu_kernel(const float* __restrict__ H,
                              const float* __restrict__ gsum, const float* __restrict__ gsq,
                              const float* __restrict__ gamma, const float* __restrict__ beta,
                              void* __restrict__ out, int nmask, float invM) {
  size_t idx = (size_t)blockIdx.x * blockDim.x + threadIdx.x;
  int n = (int)idx & nmask;
  float mean = gsum[n] * invM;
  float var  = gsq[n] * invM - mean * mean;
  float sc   = gamma[n] * rsqrtf(var + 1e-5f);
  float v    = (H[idx] - mean) * sc + beta[n];
  v = fmaxf(v, 0.f);
  if (BF16OUT) ((unsigned short*)out)[idx] = f2bf(v);
  else         ((float*)out)[idx] = v;
}

extern "C" void kernel_launch(void* const* d_in, const int* in_sizes, int n_in,
                              void* d_out, int out_size, void* d_ws, size_t ws_size,
                              hipStream_t stream) {
  (void)in_sizes; (void)n_in; (void)out_size; (void)ws_size;
  const float* xyz1    = (const float*)d_in[0];
  const float* xyz2    = (const float*)d_in[1];
  const float* points1 = (const float*)d_in[2];
  const float* points2 = (const float*)d_in[3];
  const float* W1      = (const float*)d_in[4];
  const float* b1      = (const float*)d_in[5];
  const float* g1      = (const float*)d_in[6];
  const float* be1     = (const float*)d_in[7];
  const float* W2      = (const float*)d_in[8];
  const float* b2      = (const float*)d_in[9];
  const float* g2      = (const float*)d_in[10];
  const float* be2     = (const float*)d_in[11];

  char* ws = (char*)d_ws;
  size_t off = 0;
  auto alloc = [&](size_t bytes) -> void* {
    void* p = ws + off;
    off = (off + bytes + 255) & ~(size_t)255;
    return p;
  };
  unsigned short* xbf  = (unsigned short*)alloc((size_t)M_ * INCH_ * 2); // 100.7 MB
  float*          h    = (float*)         alloc((size_t)M_ * 256 * 4);   // 134 MB (reused for h2)
  unsigned short* a2   = (unsigned short*)alloc((size_t)M_ * 256 * 2);   // 67 MB
  unsigned short* w1bf = (unsigned short*)alloc((size_t)256 * 384 * 2);
  unsigned short* w2bf = (unsigned short*)alloc((size_t)128 * 256 * 2);
  int*            idx3 = (int*)           alloc((size_t)M_ * 3 * 4);
  float*          w3   = (float*)         alloc((size_t)M_ * 3 * 4);
  float*          stats= (float*)         alloc(768 * 4);
  float *sum1 = stats, *sq1 = stats + 256, *sum2 = stats + 512, *sq2 = stats + 640;

  (void)hipMemsetAsync(stats, 0, 768 * 4, stream);

  cvt_f32_bf16<<<(256 * 384 + 255) / 256, 256, 0, stream>>>(W1, w1bf, 256 * 384);
  cvt_f32_bf16<<<(128 * 256 + 255) / 256, 256, 0, stream>>>(W2, w2bf, 128 * 256);

  knn3_kernel<<<dim3(N1_ / 256, B_), 256, 0, stream>>>(xyz1, xyz2, idx3, w3);
  interp_concat_kernel<<<M_, 128, 0, stream>>>(points1, points2, idx3, w3, xbf);

  const float invM = 1.0f / (float)M_;

  gemm_bf16_wmma<INCH_, 256><<<dim3(M_ / 128, 256 / 64), 256, 0, stream>>>(
      xbf, w1bf, b1, h, sum1, sq1);
  bnrelu_kernel<true><<<(unsigned)((size_t)M_ * 256 / 256), 256, 0, stream>>>(
      h, sum1, sq1, g1, be1, a2, 255, invM);

  gemm_bf16_wmma<256, 128><<<dim3(M_ / 128, 128 / 64), 256, 0, stream>>>(
      a2, w2bf, b2, h, sum2, sq2);
  bnrelu_kernel<false><<<(unsigned)((size_t)M_ * 128 / 256), 256, 0, stream>>>(
      h, sum2, sq2, g2, be2, d_out, 127, invM);
}
